// ConvBnReLU3DSparseFactorize_52149492908563
// MI455X (gfx1250) — compile-verified, matrix operand order checked by reading, not run on hardware
//
#include <hip/hip_runtime.h>

typedef __attribute__((ext_vector_type(16))) __bf16   v16bf;
typedef __attribute__((ext_vector_type(8)))  float    v8f;

#define CH        64
#define KW        3
#define KDIM      (KW * CH)     // 192
#define BROWS     128           // voxels per block
#define SA_STRIDE 200           // padded bf16 row stride (100 dwords -> no bank conflicts)
#define NTHREADS  256
#define GPAIRS    (NTHREADS >> 3)              // 32 (row,j) pairs staged per pass
#define GITERS    ((BROWS * KW) / GPAIRS)      // 12 async copies per thread

// One factorized sparse-conv stage: out[n, :] = sum_j in[nbr[n,j], :] @ W[j]
// in/out are bf16 with a zero sentinel row at index n.
__global__ __launch_bounds__(NTHREADS)
void conv_wmma(const __bf16* __restrict__ in,    // [(n+1)*CH], row n == zeros
               const int*    __restrict__ nbr,   // [n*KW]
               const float*  __restrict__ W,     // [KW*CH*CH] (j, cin, cout)
               __bf16*       __restrict__ out,   // [(n+1)*CH]
               int n)
{
    __shared__ __bf16 sA[BROWS * SA_STRIDE];   // gathered activations [row][k]
    __shared__ __bf16 sB[CH * SA_STRIDE];      // weights transposed   [col][k]

    const int tid  = threadIdx.x;
    const int base = blockIdx.x * BROWS;

    // ---- stage gathered activations via async global->LDS (no VGPR round-trip) ----
    // 8 threads per (row, j) pair, each lane moves 16B; all copies in flight on ASYNCcnt.
    {
        const int sub = tid & 7;
#pragma unroll
        for (int it = 0; it < GITERS; ++it) {
            const int p   = (tid >> 3) + it * GPAIRS;
            const int r   = p / KW;
            const int j   = p % KW;
            const int row = base + r;
            const int idx = (row < n) ? nbr[row * KW + j] : n;  // sentinel -> zero row
            const unsigned long long ga =
                (unsigned long long)(uintptr_t)(in + (size_t)idx * CH + sub * 8);
            const unsigned lds =
                (unsigned)(uintptr_t)(&sA[r * SA_STRIDE + j * CH + sub * 8]);
            asm volatile("global_load_async_to_lds_b128 %0, %1, off"
                         :: "v"(lds), "v"(ga) : "memory");
        }
    }

    // ---- stage weights transposed: sB[col][j*CH+c] = bf16(W[j][c][col]) ----
    for (int t = tid; t < KDIM * CH; t += NTHREADS) {
        const int col = t & (CH - 1);
        const int k   = t >> 6;             // j*CH + c
        sB[col * SA_STRIDE + k] = (__bf16)W[t];
    }

    asm volatile("s_wait_asynccnt 0x0" ::: "memory");
    __syncthreads();

    // ---- WMMA: each wave computes a 16-row x 64-col tile ----
    const int wave    = tid >> 5;   // 0..7
    const int lane    = tid & 31;
    const int half    = lane >> 4;
    const int m       = lane & 15;
    const int rowBase = wave * 16;

    // A fragments: 6 chunks of K=32, ISA 16-bit A layout
    // (VGPR v, lane half h -> k = 16*(v/4) + 8h + 2*(v&3))
    v16bf a[6];
#pragma unroll
    for (int kc = 0; kc < 6; ++kc) {
        const __bf16* sp = &sA[(rowBase + m) * SA_STRIDE + kc * 32];
#pragma unroll
        for (int v = 0; v < 8; ++v) {
            const int k0 = ((v >> 2) << 4) + (half << 3) + ((v & 3) << 1);
            a[kc][2 * v]     = sp[k0];
            a[kc][2 * v + 1] = sp[k0 + 1];
        }
    }

#pragma unroll
    for (int ct = 0; ct < 4; ++ct) {
        v8f acc = {};
#pragma unroll
        for (int kc = 0; kc < 6; ++kc) {
            v16bf b;
            const __bf16* sp = &sB[(ct * 16 + m) * SA_STRIDE + kc * 32];
#pragma unroll
            for (int v = 0; v < 8; ++v) {
                const int k0 = ((v >> 2) << 4) + (half << 3) + ((v & 3) << 1);
                b[2 * v]     = sp[k0];
                b[2 * v + 1] = sp[k0 + 1];
            }
            acc = __builtin_amdgcn_wmma_f32_16x16x32_bf16(
                false, a[kc], false, b, (short)0, acc, false, false);
        }
        // D layout: lane L, VGPR r -> (M = r + 8*(L>>4), N = L&15)
#pragma unroll
        for (int r = 0; r < 8; ++r) {
            const int row = base + rowBase + r + 8 * half;
            if (row < n)
                out[(size_t)row * CH + ct * 16 + m] = (__bf16)acc[r];
        }
    }
}

__global__ void cast_feats(const float* __restrict__ f, __bf16* __restrict__ o, int n) {
    const int i     = blockIdx.x * blockDim.x + threadIdx.x;
    const int total = (n + 1) * CH;
    if (i < total)
        o[i] = (i < n * CH) ? (__bf16)f[i] : (__bf16)0.0f;
}

__global__ void init_aux(__bf16* b1, __bf16* b2, float* stats, int n) {
    const int i = threadIdx.x;
    if (i < CH) {
        b1[(size_t)n * CH + i] = (__bf16)0.0f;
        b2[(size_t)n * CH + i] = (__bf16)0.0f;
    }
    if (i < 2 * CH) stats[i] = 0.0f;
}

__global__ __launch_bounds__(256)
void reduce_stats(const __bf16* __restrict__ h, float* __restrict__ stats, int n) {
    __shared__ float red[2][256];
    const int tid = threadIdx.x;
    const int c   = tid & 63;
    const int g   = tid >> 6;   // 0..3 row groups
    float s = 0.0f, s2 = 0.0f;
    for (int r = blockIdx.x * 4 + g; r < n; r += gridDim.x * 4) {
        const float v = (float)h[(size_t)r * CH + c];
        s  += v;
        s2 += v * v;
    }
    red[0][tid] = s;
    red[1][tid] = s2;
    __syncthreads();
    if (tid < CH) {
        float t0 = red[0][tid] + red[0][tid + 64] + red[0][tid + 128] + red[0][tid + 192];
        float t1 = red[1][tid] + red[1][tid + 64] + red[1][tid + 128] + red[1][tid + 192];
        atomicAdd(&stats[tid], t0);
        atomicAdd(&stats[CH + tid], t1);
    }
}

__global__ void bn_relu(const __bf16* __restrict__ h, const float* __restrict__ stats,
                        const float* __restrict__ gamma, const float* __restrict__ beta,
                        float* __restrict__ out, int n) {
    const int i = blockIdx.x * blockDim.x + threadIdx.x;
    if (i >= n * CH) return;
    const int   c     = i & 63;
    const float inv_n = 1.0f / (float)n;
    const float mu    = stats[c] * inv_n;
    const float var   = stats[CH + c] * inv_n - mu * mu;
    const float v = ((float)h[i] - mu) * rsqrtf(var + 1e-5f) * gamma[c] + beta[c];
    out[i] = v > 0.0f ? v : 0.0f;
}

extern "C" void kernel_launch(void* const* d_in, const int* in_sizes, int n_in,
                              void* d_out, int out_size, void* d_ws, size_t ws_size,
                              hipStream_t stream) {
    const float* feats = (const float*)d_in[0];
    const float* W1    = (const float*)d_in[1];
    const float* W2    = (const float*)d_in[2];
    const float* W3    = (const float*)d_in[3];
    const float* gamma = (const float*)d_in[4];
    const float* beta  = (const float*)d_in[5];
    const int*   nbr_z = (const int*)d_in[6];
    const int*   nbr_y = (const int*)d_in[7];
    const int*   nbr_x = (const int*)d_in[8];
    const int    n     = in_sizes[0] / CH;          // 300000

    const size_t bufElems = (size_t)(n + 1) * CH;
    __bf16* bufA  = (__bf16*)d_ws;                  // feats bf16 / h3
    __bf16* buf1  = bufA + bufElems;                // h1
    __bf16* buf2  = buf1 + bufElems;                // h2
    float*  stats = (float*)(buf2 + bufElems);      // sum[64], sumsq[64]

    init_aux<<<1, 256, 0, stream>>>(buf1, buf2, stats, n);

    const int total = (n + 1) * CH;
    cast_feats<<<(total + 255) / 256, 256, 0, stream>>>(feats, bufA, n);

    const int nblk = (n + BROWS - 1) / BROWS;
    conv_wmma<<<nblk, NTHREADS, 0, stream>>>(bufA, nbr_z, W1, buf1, n);   // (1,1,3)
    conv_wmma<<<nblk, NTHREADS, 0, stream>>>(buf1, nbr_y, W2, buf2, n);   // (1,3,1)
    conv_wmma<<<nblk, NTHREADS, 0, stream>>>(buf2, nbr_x, W3, bufA, n);   // (3,1,1)

    reduce_stats<<<1024, 256, 0, stream>>>(bufA, stats, n);
    bn_relu<<<(n * CH + 255) / 256, 256, 0, stream>>>(bufA, stats, gamma, beta,
                                                      (float*)d_out, n);
}
